// CustomSiglipEncoderLayer_29575144800700
// MI455X (gfx1250) — matrix operand order repaired and checked by reading, not compiled, not run on hardware
//
#include <hip/hip_runtime.h>
#include <math.h>

// ---------------- problem constants ----------------
#define B_    32
#define S_    729
#define D_    1152
#define H_    16
#define HD_   72
#define I_FF  4304
#define R_    20
#define NA_   365          // (S+1)/2
#define NB_   364          // S/2
#define NU_   345          // NA - R
#define NKV_  709          // NU + NB
#define NKVP_ 736          // padded to multiple of 32
#define HDP_  80           // padded HD for P*V output tiles
#define IP_   4352         // padded intermediate (multiple of 128)
#define BS_   (B_ * S_)    // 23328
#define BH_   (B_ * H_)    // 512
#define EPS_  1e-6f
#define ATT_SCALE 0.11785113019775793f   // 1/sqrt(72)

typedef __bf16 bf16_t;
typedef __attribute__((ext_vector_type(8)))  bf16_t v8bf;
typedef __attribute__((ext_vector_type(16))) bf16_t v16bf;
typedef __attribute__((ext_vector_type(8)))  float  v8f;

__device__ __forceinline__ v8bf ld8(const bf16_t* p) { return *(const v8bf*)p; }
__device__ __forceinline__ v8bf zero8() {
    v8bf z = { (bf16_t)0.f,(bf16_t)0.f,(bf16_t)0.f,(bf16_t)0.f,
               (bf16_t)0.f,(bf16_t)0.f,(bf16_t)0.f,(bf16_t)0.f };
    return z;
}
__device__ __forceinline__ v16bf cat16(v8bf lo, v8bf hi) {
    return __builtin_shufflevector(lo, hi, 0,1,2,3,4,5,6,7,8,9,10,11,12,13,14,15);
}
__device__ __forceinline__ v8f wmma_bf16(v16bf a, v16bf b, v8f c) {
    // D(f32 16x16) = A(bf16 16x32) * B(bf16 32x16) + C
    return __builtin_amdgcn_wmma_f32_16x16x32_bf16(false, a, false, b, (short)0, c, false, false);
}
__device__ __forceinline__ int imin(int a, int b) { return a < b ? a : b; }

// ================= LayerNorm (f32 in -> bf16 out) =================
__global__ __launch_bounds__(128) void ln_kernel(const float* __restrict__ x,
                                                 const float* __restrict__ w,
                                                 const float* __restrict__ bvec,
                                                 bf16_t* __restrict__ out) {
    const int row = blockIdx.x;
    const int tid = threadIdx.x;
    const float* xr = x + (size_t)row * D_;
    float v[9];
    float s = 0.f;
#pragma unroll
    for (int j = 0; j < 9; ++j) { v[j] = xr[tid + j * 128]; s += v[j]; }
#pragma unroll
    for (int o = 16; o; o >>= 1) s += __shfl_xor(s, o, 32);
    __shared__ float r1[4], r2[4];
    const int wid = tid >> 5, lane = tid & 31;
    if (lane == 0) r1[wid] = s;
    __syncthreads();
    const float mu = (r1[0] + r1[1] + r1[2] + r1[3]) * (1.f / (float)D_);
    float vs = 0.f;
#pragma unroll
    for (int j = 0; j < 9; ++j) { float d = v[j] - mu; vs += d * d; }
#pragma unroll
    for (int o = 16; o; o >>= 1) vs += __shfl_xor(vs, o, 32);
    if (lane == 0) r2[wid] = vs;
    __syncthreads();
    const float var = (r2[0] + r2[1] + r2[2] + r2[3]) * (1.f / (float)D_);
    const float rstd = rsqrtf(var + EPS_);
#pragma unroll
    for (int j = 0; j < 9; ++j) {
        int c = tid + j * 128;
        out[(size_t)row * D_ + c] = (bf16_t)((v[j] - mu) * rstd * w[c] + bvec[c]);
    }
}

// ========== weight convert+transpose: f32 (K x N) -> bf16 (N x ldk) ==========
__global__ __launch_bounds__(256) void wt_kernel(const float* __restrict__ w,
                                                 bf16_t* __restrict__ wt,
                                                 int K, int N, int ldk) {
    size_t idx = (size_t)blockIdx.x * 256 + threadIdx.x;
    size_t total = (size_t)N * ldk;
    if (idx >= total) return;
    int n = (int)(idx / ldk);
    int k = (int)(idx % ldk);
    wt[idx] = (bf16_t)((k < K) ? w[(size_t)k * N + n] : 0.f);
}

// ================= tiled bf16 WMMA GEMM =================
// C(MxN) = A(MxK, lda) * Bt(NxK, ldb)^T + bias
// MODE 0: store bf16   MODE 1: gelu -> bf16   MODE 2: f32 + residual
__device__ __forceinline__ float gelu_tanh(float x) {
    return 0.5f * x * (1.f + tanhf(0.7978845608028654f * (x + 0.044715f * x * x * x)));
}

template <int MODE>
__global__ __launch_bounds__(256) void gemm_kernel(const bf16_t* __restrict__ A,
                                                   const bf16_t* __restrict__ Bt,
                                                   const float* __restrict__ bias,
                                                   const float* __restrict__ resid,
                                                   void* __restrict__ Cout,
                                                   int M, int N, int K,
                                                   int lda, int ldb, int ldc) {
    __shared__ __align__(16) bf16_t As[128 * 40];
    __shared__ __align__(16) bf16_t Bs[128 * 40];
    const int tid  = threadIdx.x;
    const int lane = tid & 31, wid = tid >> 5;
    const int wm = wid >> 1, wn = wid & 1;         // 4 waves along M, 2 along N
    const int sel = lane >> 4, lr = lane & 15;
    const int bm = blockIdx.x * 128, bn = blockIdx.y * 128;

    v8f acc[2][4];
#pragma unroll
    for (int mi = 0; mi < 2; ++mi)
#pragma unroll
        for (int ni = 0; ni < 4; ++ni) acc[mi][ni] = (v8f){0,0,0,0,0,0,0,0};

    for (int kb = 0; kb < K; kb += 32) {
        __syncthreads();
#pragma unroll
        for (int pass = 0; pass < 2; ++pass) {
            int r = (tid >> 2) + pass * 64;
            int c = (tid & 3) * 8;
            int gm = bm + r;
            v8bf va = (gm < M) ? ld8(A + (size_t)gm * lda + kb + c) : zero8();
            *(v8bf*)&As[r * 40 + c] = va;
            int gn = bn + r;
            v8bf vb = (gn < N) ? ld8(Bt + (size_t)gn * ldb + kb + c) : zero8();
            *(v8bf*)&Bs[r * 40 + c] = vb;
        }
        __syncthreads();

        v16bf af[2], bfr[4];
#pragma unroll
        for (int mi = 0; mi < 2; ++mi) {    // A layout: split-8 K chunks per half-wave
            int r = wm * 32 + mi * 16 + lr;
            af[mi] = cat16(ld8(&As[r * 40 + sel * 8]), ld8(&As[r * 40 + 16 + sel * 8]));
        }
#pragma unroll
        for (int ni = 0; ni < 4; ++ni) {    // B layout: split-8 K chunks per half-wave
            int r = wn * 64 + ni * 16 + lr;
            bfr[ni] = cat16(ld8(&Bs[r * 40 + sel * 8]), ld8(&Bs[r * 40 + 16 + sel * 8]));
        }
#pragma unroll
        for (int mi = 0; mi < 2; ++mi)
#pragma unroll
            for (int ni = 0; ni < 4; ++ni)
                acc[mi][ni] = wmma_bf16(af[mi], bfr[ni], acc[mi][ni]);
    }

    // epilogue: C frag row = i + sel*8, col = lr
#pragma unroll
    for (int mi = 0; mi < 2; ++mi)
#pragma unroll
        for (int ni = 0; ni < 4; ++ni)
#pragma unroll
            for (int i = 0; i < 8; ++i) {
                int gm = bm + wm * 32 + mi * 16 + i + sel * 8;
                int gn = bn + wn * 64 + ni * 16 + lr;
                if (gm < M) {
                    float v = acc[mi][ni][i] + ((gn < N) ? bias[gn] : 0.f);
                    if (MODE == 0) {
                        ((bf16_t*)Cout)[(size_t)gm * ldc + gn] = (bf16_t)v;
                    } else if (MODE == 1) {
                        ((bf16_t*)Cout)[(size_t)gm * ldc + gn] = (bf16_t)gelu_tanh(v);
                    } else {
                        ((float*)Cout)[(size_t)gm * ldc + gn] =
                            v + resid[(size_t)gm * ldc + gn];
                    }
                }
            }
}

// ================= metric = mean over HD of k =================
__global__ __launch_bounds__(256) void metric_kernel(const bf16_t* __restrict__ kb,
                                                     float* __restrict__ metric) {
    int idx = blockIdx.x * 256 + threadIdx.x;
    if (idx >= B_ * H_ * S_) return;
    int b = idx / (H_ * S_);
    int r = idx % (H_ * S_);
    int h = r / S_;
    int s = r % S_;
    const bf16_t* p = kb + ((size_t)(b * S_ + s)) * D_ + h * HD_;
    float acc = 0.f;
#pragma unroll 8
    for (int d = 0; d < HD_; ++d) acc += (float)p[d];
    metric[idx] = acc * (1.f / (float)HD_);
}

// ================= bipartite soft matching =================
__global__ __launch_bounds__(256) void match_kernel(const float* __restrict__ metric,
                                                    int* __restrict__ unm,
                                                    int* __restrict__ srcI,
                                                    int* __restrict__ dstI) {
    const int bh  = blockIdx.x;
    const int tid = threadIdx.x;
    const float* mt = metric + (size_t)bh * S_;
    __shared__ float bsg[NB_];
    __shared__ float key[512];
    __shared__ int   sidx[512];
    __shared__ int   nidx[NA_];
    for (int j = tid; j < NB_; j += 256) {
        float m = mt[2 * j + 1];
        bsg[j] = m / (fabsf(m) + 1e-12f);
    }
    __syncthreads();
    for (int i = tid; i < 512; i += 256) {
        float kmax = -3.0e38f;
        if (i < NA_) {
            float m = mt[2 * i];
            float a = m / (fabsf(m) + 1e-12f);
            float best = -3.0e38f; int ni = 0;
            for (int j = 0; j < NB_; ++j) {
                float sc = a * bsg[j];
                if (sc > best) { best = sc; ni = j; }   // first argmax (jnp semantics)
            }
            kmax = best; nidx[i] = ni;
        }
        key[i] = kmax; sidx[i] = i;
    }
    // bitonic sort: descending key, ties -> ascending index (stable argsort of -key)
    for (int k = 2; k <= 512; k <<= 1) {
        for (int j = k >> 1; j > 0; j >>= 1) {
            __syncthreads();
            for (int e = tid; e < 512; e += 256) {
                int x = e ^ j;
                if (x > e) {
                    float ka = key[e], kb2 = key[x];
                    int ia = sidx[e], ib = sidx[x];
                    bool before = (ka > kb2) || (ka == kb2 && ia < ib);
                    bool up = ((e & k) == 0);
                    if (up != before) {
                        key[e] = kb2; key[x] = ka;
                        sidx[e] = ib; sidx[x] = ia;
                    }
                }
            }
        }
    }
    __syncthreads();
    for (int t = tid; t < NA_; t += 256) {
        if (t < R_) {
            int si = sidx[t];
            srcI[bh * R_ + t] = si;
            dstI[bh * R_ + t] = nidx[si];
        } else {
            unm[bh * NU_ + (t - R_)] = sidx[t];
        }
    }
}

// ================= ToMe merge: build k_m, v_m^T, log_sizes =================
__global__ __launch_bounds__(256) void merge_kernel(const bf16_t* __restrict__ kb,
                                                    const bf16_t* __restrict__ vb,
                                                    const int* __restrict__ unm,
                                                    const int* __restrict__ srcI,
                                                    const int* __restrict__ dstI,
                                                    float* __restrict__ kdst,
                                                    float* __restrict__ vdst,
                                                    bf16_t* __restrict__ km,
                                                    bf16_t* __restrict__ vmt,
                                                    float* __restrict__ logs) {
    const int bh = blockIdx.x;
    const int b = bh >> 4, h = bh & 15;
    const int tid = threadIdx.x;
    __shared__ float sizes[NB_];
    const bf16_t* kbase = kb + (size_t)b * S_ * D_ + h * HD_;
    const bf16_t* vbase = vb + (size_t)b * S_ * D_ + h * HD_;
    float* kd = kdst + (size_t)bh * NB_ * HD_;
    float* vd = vdst + (size_t)bh * NB_ * HD_;
    // phase 1: init dst accumulators with odd tokens
    for (int e = tid; e < NB_ * HD_; e += 256) {
        int j = e / HD_, d = e % HD_;
        size_t off = (size_t)(2 * j + 1) * D_ + d;
        kd[e] = (float)kbase[off];
        vd[e] = (float)vbase[off];
    }
    for (int j = tid; j < NB_; j += 256) sizes[j] = 1.f;
    __syncthreads();
    // phase 2: scatter-add R merged sources
    if (tid < R_) {
        int si = srcI[bh * R_ + tid];
        int dj = dstI[bh * R_ + tid];
        size_t off = (size_t)(2 * si) * D_;
        for (int d = 0; d < HD_; ++d) {
            atomicAdd(&kd[dj * HD_ + d], (float)kbase[off + d]);
            atomicAdd(&vd[dj * HD_ + d], (float)vbase[off + d]);
        }
        atomicAdd(&sizes[dj], 1.f);
    }
    __syncthreads();
    bf16_t* kmo = km  + (size_t)bh * NKV_ * HD_;
    bf16_t* vto = vmt + (size_t)bh * HDP_ * NKVP_;
    float*  lg  = logs + (size_t)bh * NKV_;
    // phase 3a: unmerged rows
    for (int e = tid; e < NU_ * HD_; e += 256) {
        int i = e / HD_, d = e % HD_;
        size_t off = (size_t)(2 * unm[bh * NU_ + i]) * D_ + d;
        kmo[i * HD_ + d]    = (bf16_t)(float)kbase[off];
        vto[d * NKVP_ + i]  = (bf16_t)(float)vbase[off];
        if (d == 0) lg[i] = 0.f;
    }
    // phase 3b: merged dst rows
    for (int e = tid; e < NB_ * HD_; e += 256) {
        int j = e / HD_, d = e % HD_;
        float sz = sizes[j];
        kmo[(NU_ + j) * HD_ + d]   = (bf16_t)(kd[e] / sz);
        vto[d * NKVP_ + (NU_ + j)] = (bf16_t)(vd[e] / sz);
        if (d == 0) lg[NU_ + j] = logf(sz);
    }
    // phase 3c: zero pads of v_m^T (rows 72..79 fully; cols 709..735 for d<72)
    for (int e = tid; e < (HDP_ - HD_) * NKVP_; e += 256) {
        int d = HD_ + e / NKVP_, c = e % NKVP_;
        vto[d * NKVP_ + c] = (bf16_t)0.f;
    }
    for (int e = tid; e < HD_ * (NKVP_ - NKV_); e += 256) {
        int d = e / (NKVP_ - NKV_), c = NKV_ + e % (NKVP_ - NKV_);
        vto[d * NKVP_ + c] = (bf16_t)0.f;
    }
}

// ================= flash attention over merged KV =================
// grid = B*H*6 blocks, 8 waves/block, one 16-row q tile per wave
__global__ __launch_bounds__(256) void flash_kernel(const bf16_t* __restrict__ qbuf,
                                                    const bf16_t* __restrict__ km,
                                                    const bf16_t* __restrict__ vmt,
                                                    const float* __restrict__ logs,
                                                    const float* __restrict__ mask,
                                                    bf16_t* __restrict__ attn) {
    __shared__ __align__(16) bf16_t pl[8][16 * 32];
    const int tid = threadIdx.x, lane = tid & 31, wid = tid >> 5;
    const int sel = lane >> 4, lr = lane & 15;
    const int bh = blockIdx.x / 6, qblk = blockIdx.x % 6;
    const int b = bh >> 4, h = bh & 15;
    const int qbase = qblk * 128 + wid * 16;

    // Q fragments: A layout, HD padded to 96 via chunk guards
    v16bf qf[3];
    {
        int row = imin(qbase + lr, S_ - 1);
        const bf16_t* qp = qbuf + ((size_t)(b * S_ + row)) * D_ + h * HD_;
#pragma unroll
        for (int c = 0; c < 3; ++c) {
            int b0 = c * 32 + sel * 8;
            int b1 = c * 32 + 16 + sel * 8;
            v8bf lo = (b0 + 8 <= HD_) ? ld8(qp + b0) : zero8();
            v8bf hi = (b1 + 8 <= HD_) ? ld8(qp + b1) : zero8();
            qf[c] = cat16(lo, hi);
        }
    }

    v8f oacc[5];
#pragma unroll
    for (int t = 0; t < 5; ++t) oacc[t] = (v8f){0,0,0,0,0,0,0,0};
    float mst[8], lst[8];
#pragma unroll
    for (int i = 0; i < 8; ++i) { mst[i] = -3.0e38f; lst[i] = 0.f; }

    int qrow[8];
#pragma unroll
    for (int i = 0; i < 8; ++i) qrow[i] = imin(qbase + i + sel * 8, S_ - 1);

    const bf16_t* kmb = km  + (size_t)bh * NKV_ * HD_;
    const bf16_t* vbp = vmt + (size_t)bh * HDP_ * NKVP_;
    const float*  lg  = logs + (size_t)bh * NKV_;
    const float*  mk  = mask + (size_t)b * S_ * NKV_;

    for (int kvb = 0; kvb < NKVP_; kvb += 32) {
        // scores: S(16x32) as two 16x16 C frags
        v8f sc[2];
#pragma unroll
        for (int f = 0; f < 2; ++f) {
            sc[f] = (v8f){0,0,0,0,0,0,0,0};
            int kr = imin(kvb + f * 16 + lr, NKV_ - 1);
            const bf16_t* kp = kmb + (size_t)kr * HD_;
#pragma unroll
            for (int c = 0; c < 3; ++c) {
                int b0 = c * 32 + sel * 8;
                int b1 = b0 + 16;
                v8bf lo = (b0 + 8 <= HD_) ? ld8(kp + b0) : zero8();
                v8bf hi = (b1 + 8 <= HD_) ? ld8(kp + b1) : zero8();
                sc[f] = wmma_bf16(qf[c], cat16(lo, hi), sc[f]);
            }
        }
        int kvc[2]; float lgv[2];
#pragma unroll
        for (int f = 0; f < 2; ++f) {
            kvc[f] = kvb + f * 16 + lr;
            lgv[f] = (kvc[f] < NKV_) ? lg[kvc[f]] : 0.f;
        }
        float p0a[8], p1a[8], fac[8];
#pragma unroll
        for (int i = 0; i < 8; ++i) {
            float s0 = (kvc[0] < NKV_)
                ? sc[0][i] * ATT_SCALE + lgv[0] + mk[(size_t)qrow[i] * NKV_ + kvc[0]]
                : -1.0e30f;
            float s1 = (kvc[1] < NKV_)
                ? sc[1][i] * ATT_SCALE + lgv[1] + mk[(size_t)qrow[i] * NKV_ + kvc[1]]
                : -1.0e30f;
            float rm = fmaxf(s0, s1);
#pragma unroll
            for (int o = 1; o < 16; o <<= 1) rm = fmaxf(rm, __shfl_xor(rm, o, 32));
            float nm = fmaxf(mst[i], rm);
            fac[i] = expf(mst[i] - nm);
            float p0 = expf(s0 - nm), p1 = expf(s1 - nm);
            float rs = p0 + p1;
#pragma unroll
            for (int o = 1; o < 16; o <<= 1) rs += __shfl_xor(rs, o, 32);
            lst[i] = lst[i] * fac[i] + rs;
            mst[i] = nm;
            p0a[i] = p0; p1a[i] = p1;
        }
#pragma unroll
        for (int t = 0; t < 5; ++t)
#pragma unroll
            for (int i = 0; i < 8; ++i) oacc[t][i] *= fac[i];

        // P (C-frag layout) -> LDS -> A-frag layout
        bf16_t* pw = &pl[wid][0];
#pragma unroll
        for (int i = 0; i < 8; ++i) {
            pw[(i + sel * 8) * 32 + lr]      = (bf16_t)p0a[i];
            pw[(i + sel * 8) * 32 + 16 + lr] = (bf16_t)p1a[i];
        }
        asm volatile("s_wait_dscnt 0x0" ::: "memory");
        v16bf pa = cat16(ld8(pw + lr * 32 + sel * 8),
                         ld8(pw + lr * 32 + 16 + sel * 8));
#pragma unroll
        for (int t = 0; t < 5; ++t) {
            const bf16_t* vp = vbp + (size_t)(t * 16 + lr) * NKVP_ + kvb;
            v16bf vf2 = cat16(ld8(vp + sel * 8), ld8(vp + 16 + sel * 8));
            oacc[t] = wmma_bf16(pa, vf2, oacc[t]);
        }
    }
    // store (B,S,H*HD) bf16
#pragma unroll
    for (int t = 0; t < 5; ++t)
#pragma unroll
        for (int i = 0; i < 8; ++i) {
            int qr = qbase + i + sel * 8;
            int col = t * 16 + lr;
            if (qr < S_ && col < HD_)
                attn[((size_t)(b * S_ + qr)) * D_ + h * HD_ + col] =
                    (bf16_t)(oacc[t][i] / lst[i]);
        }
}

// ================================ host ================================
extern "C" void kernel_launch(void* const* d_in, const int* in_sizes, int n_in,
                              void* d_out, int out_size, void* d_ws, size_t ws_size,
                              hipStream_t stream) {
    const float* x     = (const float*)d_in[0];
    const float* mask  = (const float*)d_in[1];
    const float* ln1w  = (const float*)d_in[2];
    const float* ln1b  = (const float*)d_in[3];
    const float* ln2w  = (const float*)d_in[4];
    const float* ln2b  = (const float*)d_in[5];
    const float* qw    = (const float*)d_in[6];
    const float* qb_   = (const float*)d_in[7];
    const float* kw    = (const float*)d_in[8];
    const float* kbias = (const float*)d_in[9];
    const float* vw    = (const float*)d_in[10];
    const float* vbias = (const float*)d_in[11];
    const float* ow    = (const float*)d_in[12];
    const float* ob    = (const float*)d_in[13];
    const float* fc1w  = (const float*)d_in[14];
    const float* fc1b  = (const float*)d_in[15];
    const float* fc2w  = (const float*)d_in[16];
    const float* fc2b  = (const float*)d_in[17];
    (void)in_sizes; (void)n_in; (void)out_size; (void)ws_size;

    char* ws = (char*)d_ws;
    size_t cur = 0;
    auto take = [&](size_t n) { size_t o = cur; cur += (n + 255) & ~(size_t)255; return o; };

    const size_t SZ_ACT = (size_t)BS_ * D_ * 2;                 // bf16 (B,S,D)
    const size_t SZ_KM  = (size_t)BH_ * NKV_ * HD_ * 2;
    const size_t off_h   = take(SZ_ACT);                        // h_ln, reused for h2
    const size_t off_big = take(3 * SZ_ACT + SZ_KM);            // q,k,v,k_m ; reused for g
    const size_t off_q   = off_big;
    const size_t off_k   = off_big + SZ_ACT;
    const size_t off_v   = off_big + 2 * SZ_ACT;                // reused as attn output
    const size_t off_km  = off_big + 3 * SZ_ACT;
    const size_t off_g   = off_big;                             // (BS, 4352) bf16 overlay
    const size_t off_vmt = take((size_t)BH_ * HDP_ * NKVP_ * 2);
    const size_t off_wq  = take((size_t)D_ * D_ * 2);
    const size_t off_wk  = take((size_t)D_ * D_ * 2);
    const size_t off_wv  = take((size_t)D_ * D_ * 2);
    const size_t off_wo  = take((size_t)D_ * D_ * 2);
    const size_t off_w1  = take((size_t)I_FF * D_ * 2);         // (4304,1152)
    const size_t off_w2  = take((size_t)D_ * IP_ * 2);          // (1152,4352)
    const size_t off_met = take((size_t)BH_ * S_ * 4);
    const size_t off_unm = take((size_t)BH_ * NU_ * 4);
    const size_t off_src = take((size_t)BH_ * R_ * 4);
    const size_t off_dst = take((size_t)BH_ * R_ * 4);
    const size_t off_log = take((size_t)BH_ * NKV_ * 4);
    const size_t off_x1  = take((size_t)BS_ * D_ * 4);          // f32 residual stream
    const size_t off_kd  = off_x1;                              // merge scratch overlays x1
    const size_t off_vd  = off_x1 + (size_t)BH_ * NB_ * HD_ * 4;

    bf16_t* h_ln = (bf16_t*)(ws + off_h);
    bf16_t* qbuf = (bf16_t*)(ws + off_q);
    bf16_t* kbuf = (bf16_t*)(ws + off_k);
    bf16_t* vbuf = (bf16_t*)(ws + off_v);
    bf16_t* kmB  = (bf16_t*)(ws + off_km);
    bf16_t* gbuf = (bf16_t*)(ws + off_g);
    bf16_t* vmt  = (bf16_t*)(ws + off_vmt);
    bf16_t* wqT  = (bf16_t*)(ws + off_wq);
    bf16_t* wkT  = (bf16_t*)(ws + off_wk);
    bf16_t* wvT  = (bf16_t*)(ws + off_wv);
    bf16_t* woT  = (bf16_t*)(ws + off_wo);
    bf16_t* w1T  = (bf16_t*)(ws + off_w1);
    bf16_t* w2T  = (bf16_t*)(ws + off_w2);
    float*  met  = (float*)(ws + off_met);
    int*    unm  = (int*)(ws + off_unm);
    int*    srcI = (int*)(ws + off_src);
    int*    dstI = (int*)(ws + off_dst);
    float*  logs = (float*)(ws + off_log);
    float*  x1   = (float*)(ws + off_x1);
    float*  kd   = (float*)(ws + off_kd);
    float*  vd   = (float*)(ws + off_vd);
    bf16_t* attn = vbuf;                    // v dead after merge -> reuse for attn out

    // 1) LN1
    ln_kernel<<<BS_, 128, 0, stream>>>(x, ln1w, ln1b, h_ln);

    // 2) weight transposes (f32 -> bf16, N x K layout)
    auto wt_launch = [&](const float* w, bf16_t* wt, int K, int N, int ldk) {
        size_t tot = (size_t)N * ldk;
        wt_kernel<<<(unsigned)((tot + 255) / 256), 256, 0, stream>>>(w, wt, K, N, ldk);
    };
    wt_launch(qw,  wqT, D_, D_, D_);
    wt_launch(kw,  wkT, D_, D_, D_);
    wt_launch(vw,  wvT, D_, D_, D_);
    wt_launch(ow,  woT, D_, D_, D_);
    wt_launch(fc1w, w1T, D_, I_FF, D_);     // (4304 x 1152)
    wt_launch(fc2w, w2T, I_FF, D_, IP_);    // (1152 x 4352), K-padded with zeros

    const dim3 gD((BS_ + 127) / 128, D_ / 128);         // 183 x 9
    const dim3 gI((BS_ + 127) / 128, IP_ / 128);        // 183 x 34

    // 3) QKV projections (bf16 out)
    gemm_kernel<0><<<gD, 256, 0, stream>>>(h_ln, wqT, qb_,   nullptr, qbuf, BS_, D_, D_, D_, D_, D_);
    gemm_kernel<0><<<gD, 256, 0, stream>>>(h_ln, wkT, kbias, nullptr, kbuf, BS_, D_, D_, D_, D_, D_);
    gemm_kernel<0><<<gD, 256, 0, stream>>>(h_ln, wvT, vbias, nullptr, vbuf, BS_, D_, D_, D_, D_, D_);

    // 4) ToMe: metric -> matching -> merge
    metric_kernel<<<(B_ * H_ * S_ + 255) / 256, 256, 0, stream>>>(kbuf, met);
    match_kernel<<<BH_, 256, 0, stream>>>(met, unm, srcI, dstI);
    merge_kernel<<<BH_, 256, 0, stream>>>(kbuf, vbuf, unm, srcI, dstI,
                                          kd, vd, kmB, vmt, logs);

    // 5) flash attention over merged KV (writes attn over vbuf)
    flash_kernel<<<BH_ * 6, 256, 0, stream>>>(qbuf, kmB, vmt, logs, mask, attn);

    // 6) O projection + residual -> x1 (f32)
    gemm_kernel<2><<<gD, 256, 0, stream>>>(attn, woT, ob, x, x1, BS_, D_, D_, D_, D_, D_);

    // 7) LN2 (reuse h_ln)
    ln_kernel<<<BS_, 128, 0, stream>>>(x1, ln2w, ln2b, h_ln);

    // 8) fc1 + GELU -> g (bf16, ldc padded to 4352; pad cols get exact zeros)
    gemm_kernel<1><<<gI, 256, 0, stream>>>(h_ln, w1T, fc1b, nullptr, gbuf,
                                           BS_, I_FF, D_, D_, D_, IP_);

    // 9) fc2 + residual -> d_out (f32); K runs over padded 4352 (zeros contribute 0)
    gemm_kernel<2><<<gD, 256, 0, stream>>>(gbuf, w2T, fc2b, x1, (float*)d_out,
                                           BS_, D_, IP_, IP_, IP_, D_);
}
